// SymplecticNeuralSolver_37065567764798
// MI455X (gfx1250) — compile-verified, matrix operand order checked by reading, not run on hardware
//
#include <hip/hip_runtime.h>

// Problem constants (from reference)
#define GD   64
#define NN   (GD * GD)   // 4096 sites per batch
#define BB   8
#define DD   32          // q / p dim
#define DANC 16
#define DIN  80          // 2*DD + DANC
#define KPAD 96          // DIN padded up to 3 K-tiles of 32
#define HIDN 128
#define NSTEPS 18

typedef __attribute__((ext_vector_type(16))) __bf16 v16bf;
typedef __attribute__((ext_vector_type(2)))  __bf16 v2bf;
typedef __attribute__((ext_vector_type(8)))  float  v8f;

union Frag { v16bf v; unsigned u[8]; };

__device__ __host__ __forceinline__ unsigned short f2bf_sw(float f) {
  unsigned x = __builtin_bit_cast(unsigned, f);
  unsigned r = x + 0x7FFFu + ((x >> 16) & 1u);   // round-to-nearest-even
  return (unsigned short)(r >> 16);
}

// Pack two f32 -> packed bf16 dword. Prefer HW v_cvt_pk_bf16_f32 (1 VALU).
__device__ __forceinline__ unsigned pack2(float a, float b) {
#if __has_builtin(__builtin_amdgcn_cvt_pk_bf16_f32)
  v2bf p = __builtin_amdgcn_cvt_pk_bf16_f32(a, b);
  return __builtin_bit_cast(unsigned, p);
#else
  return (unsigned)f2bf_sw(a) | ((unsigned)f2bf_sw(b) << 16);
#endif
}

__device__ __forceinline__ unsigned short bf1(float a) {
#if __has_builtin(__builtin_amdgcn_cvt_pk_bf16_f32)
  v2bf p = __builtin_amdgcn_cvt_pk_bf16_f32(a, a);
  return (unsigned short)(__builtin_bit_cast(unsigned, p) & 0xFFFFu);
#else
  return f2bf_sw(a);
#endif
}

// tanh: CDNA5 has V_TANH_F32 (listed as a TRANS op in the ISA's S_DELAY_ALU
// section); fall back to exp2/rcp rational form if the builtin is absent.
__device__ __forceinline__ float tanh_hw(float x) {
#if __has_builtin(__builtin_amdgcn_tanhf)
  return __builtin_amdgcn_tanhf(x);
#else
  float e = __builtin_amdgcn_exp2f(x * 2.8853900817779268f);
  return 1.0f - 2.0f * __builtin_amdgcn_rcpf(e + 1.0f);
#endif
}

// Compile-time memory ordering only: LDS ops from one wave execute in order
// (ISA ch.11), and all LDS regions here are wave-private, so no s_barrier or
// s_wait is needed -- just stop the compiler from reordering across this point.
__device__ __forceinline__ void wave_lds_fence() { asm volatile("" ::: "memory"); }

// ---------------------------------------------------------------------------
// Prep: split x -> q,p working buffers in d_ws
// ---------------------------------------------------------------------------
__global__ __launch_bounds__(256) void copy_in_kernel(
    const float* __restrict__ x, float* __restrict__ qbuf, float* __restrict__ pbuf) {
  int i = blockIdx.x * blockDim.x + threadIdx.x;   // over BB*NN*DD
  int site = i >> 5, d = i & 31;
  qbuf[i] = x[site * DIN + d];
  pbuf[i] = x[site * DIN + DD + d];
}

// ---------------------------------------------------------------------------
// Prep: swizzle a row-major f32 weight matrix W[K x Ncols] into packed bf16
// B-operand fragments for v_wmma_f32_16x16x32_bf16.
// Per ISA 16-bit operand layout (§7.12.2): lane = column (n), VGPR v holds
// K pair: k = kt*32 + (v<4?0:16) + (lane>=16?8:0) + (v&3)*2.
// out[gid] packs (k, k+1) as two bf16; gid = ((kt*nTiles+nt)*32+lane)*8+v
// ---------------------------------------------------------------------------
__global__ __launch_bounds__(256) void swizzle_w_kernel(
    const float* __restrict__ W, int K, int Ncols, int nTiles,
    unsigned* __restrict__ out, int total) {
  int gid = blockIdx.x * blockDim.x + threadIdx.x;
  if (gid >= total) return;
  int f   = gid >> 8;
  int rem = gid & 255;
  int ln  = rem >> 3;
  int v   = rem & 7;
  int kt  = f / nTiles, nt = f % nTiles;
  int k = kt * 32 + ((v < 4) ? 0 : 16) + ((ln >> 4) & 1) * 8 + (v & 3) * 2;
  int n = nt * 16 + (ln & 15);
  float f0 = (k     < K) ? W[(size_t)k       * Ncols + n] : 0.0f;
  float f1 = (k + 1 < K) ? W[(size_t)(k + 1) * Ncols + n] : 0.0f;
  out[gid] = pack2(f0, f1);
}

// ---------------------------------------------------------------------------
// One half-step: zdst[site] += dt(batch) * MLP(zsrc neighborhood, xi)
// One wave handles 16 sites:
//   L1: [16x96](bf16) @ [96x128](bf16) -> f32, tanh  (3 K-tiles x 8 N-tiles)
//   L2: [16x128]      @ [128x32]       -> f32        (4 K-tiles x 2 N-tiles)
// ---------------------------------------------------------------------------
__global__ __launch_bounds__(256) void half_step_kernel(
    const float* __restrict__ zsrc, float* __restrict__ zdst,
    const float* __restrict__ x, const int* __restrict__ nbrs,
    const unsigned* __restrict__ w1f, const float* __restrict__ b1,
    const unsigned* __restrict__ w2f, const float* __restrict__ b2,
    const float* __restrict__ tfin, float t_cur, float dt_max) {
  __shared__ __align__(16) unsigned short ain[8][16][KPAD];   // 24 KB
  __shared__ __align__(16) unsigned short hbuf[8][16][HIDN];  // 32 KB

  const int lane = threadIdx.x & 31;
  const int wv   = threadIdx.x >> 5;
  const int siteBase  = blockIdx.x * 128 + wv * 16;
  const int batch     = siteBase >> 12;        // NN = 4096
  const int batchBase = batch << 12;

  // ---- stage input tile as bf16: row r = [ z(32) | mean-nbr z(32) | xi(16) | 0(16) ]
  // lane = (row, column-half): no divisions, float4 global reads, u32 LDS writes.
  {
    const int r     = lane & 15;
    const int chalf = lane >> 4;
    const int site  = siteBase + r;
    const int ls    = site & (NN - 1);
    unsigned* dst = (unsigned*)&ain[wv][r][0];   // 48 uints per row

    const int4 nb = *(const int4*)&nbrs[ls * 4];
    const float4* zn0 = (const float4*)(zsrc + (size_t)(batchBase + nb.x) * DD);
    const float4* zn1 = (const float4*)(zsrc + (size_t)(batchBase + nb.y) * DD);
    const float4* zn2 = (const float4*)(zsrc + (size_t)(batchBase + nb.z) * DD);
    const float4* zn3 = (const float4*)(zsrc + (size_t)(batchBase + nb.w) * DD);

    if (chalf == 0) {
      // cols 0..31: self z
      const float4* zs = (const float4*)(zsrc + (size_t)site * DD);
#pragma unroll
      for (int j = 0; j < 8; ++j) {
        float4 v = zs[j];
        dst[j * 2 + 0] = pack2(v.x, v.y);
        dst[j * 2 + 1] = pack2(v.z, v.w);
      }
      // cols 32..47: neighbour mean, d = 0..15
#pragma unroll
      for (int j = 0; j < 4; ++j) {
        float4 a = zn0[j], b = zn1[j], c = zn2[j], d4 = zn3[j];
        float mx = 0.25f * (a.x + b.x + c.x + d4.x);
        float my = 0.25f * (a.y + b.y + c.y + d4.y);
        float mz = 0.25f * (a.z + b.z + c.z + d4.z);
        float mw = 0.25f * (a.w + b.w + c.w + d4.w);
        dst[16 + j * 2 + 0] = pack2(mx, my);
        dst[16 + j * 2 + 1] = pack2(mz, mw);
      }
    } else {
      // cols 48..63: neighbour mean, d = 16..31
#pragma unroll
      for (int j = 4; j < 8; ++j) {
        float4 a = zn0[j], b = zn1[j], c = zn2[j], d4 = zn3[j];
        float mx = 0.25f * (a.x + b.x + c.x + d4.x);
        float my = 0.25f * (a.y + b.y + c.y + d4.y);
        float mz = 0.25f * (a.z + b.z + c.z + d4.z);
        float mw = 0.25f * (a.w + b.w + c.w + d4.w);
        dst[16 + j * 2 + 0] = pack2(mx, my);
        dst[16 + j * 2 + 1] = pack2(mz, mw);
      }
      // cols 64..79: xi
      const float4* xs = (const float4*)(x + (size_t)site * DIN + 2 * DD);
#pragma unroll
      for (int j = 0; j < 4; ++j) {
        float4 v = xs[j];
        dst[32 + j * 2 + 0] = pack2(v.x, v.y);
        dst[32 + j * 2 + 1] = pack2(v.z, v.w);
      }
      // cols 80..95: zero pad
#pragma unroll
      for (int j = 40; j < 48; ++j) dst[j] = 0u;
    }
  }
  wave_lds_fence();

  // ---- layer 1 A fragments from LDS (row m = lane&15, ISA K packing)
  Frag a1[3];
  {
    const int m   = lane & 15;
    const int hi8 = ((lane >> 4) & 1) * 8;
    const unsigned* row = (const unsigned*)&ain[wv][m][0];
#pragma unroll
    for (int kt = 0; kt < 3; ++kt)
#pragma unroll
      for (int v = 0; v < 8; ++v) {
        int k = kt * 32 + ((v < 4) ? 0 : 16) + hi8 + (v & 3) * 2;
        a1[kt].u[v] = row[k >> 1];
      }
  }

#pragma unroll
  for (int nt = 0; nt < 8; ++nt) {
    v8f acc = {};
#pragma unroll
    for (int kt = 0; kt < 3; ++kt) {
      Frag b;
      const unsigned* bp = &w1f[(size_t)(((kt << 3) + nt) * 32 + lane) << 3];
#pragma unroll
      for (int v = 0; v < 8; ++v) b.u[v] = bp[v];
      acc = __builtin_amdgcn_wmma_f32_16x16x32_bf16(false, a1[kt].v, false, b.v,
                                                    (short)0, acc, false, false);
    }
    // bias + hw tanh, write hidden tile to LDS (D-layout -> row-major bf16)
    const int n  = nt * 16 + (lane & 15);
    const int mb = (lane >= 16) ? 8 : 0;
    const float bias = b1[n];
#pragma unroll
    for (int v = 0; v < 8; ++v)
      hbuf[wv][mb + v][n] = bf1(tanh_hw(acc[v] + bias));
  }
  wave_lds_fence();

  // ---- layer 2 A fragments from LDS hidden tile
  Frag a2[4];
  {
    const int m   = lane & 15;
    const int hi8 = ((lane >> 4) & 1) * 8;
    const unsigned* row = (const unsigned*)&hbuf[wv][m][0];
#pragma unroll
    for (int kt = 0; kt < 4; ++kt)
#pragma unroll
      for (int v = 0; v < 8; ++v) {
        int k = kt * 32 + ((v < 4) ? 0 : 16) + hi8 + (v & 3) * 2;
        a2[kt].u[v] = row[k >> 1];
      }
  }

  const float dtf = fminf(dt_max, fmaxf(0.0f, tfin[batch] - t_cur));

#pragma unroll
  for (int nt = 0; nt < 2; ++nt) {
    v8f acc = {};
#pragma unroll
    for (int kt = 0; kt < 4; ++kt) {
      Frag b;
      const unsigned* bp = &w2f[(size_t)(((kt << 1) + nt) * 32 + lane) << 3];
#pragma unroll
      for (int v = 0; v < 8; ++v) b.u[v] = bp[v];
      acc = __builtin_amdgcn_wmma_f32_16x16x32_bf16(false, a2[kt].v, false, b.v,
                                                    (short)0, acc, false, false);
    }
    const int d  = nt * 16 + (lane & 15);
    const int mb = (lane >= 16) ? 8 : 0;
    const float bias = b2[d];
#pragma unroll
    for (int v = 0; v < 8; ++v) {
      const int site = siteBase + mb + v;
      zdst[site * DD + d] += dtf * (acc[v] + bias);
    }
  }
}

// ---------------------------------------------------------------------------
// Assemble output [q | p | xi]
// ---------------------------------------------------------------------------
__global__ __launch_bounds__(256) void finalize_kernel(
    const float* __restrict__ qbuf, const float* __restrict__ pbuf,
    const float* __restrict__ x, float* __restrict__ out) {
  int i = blockIdx.x * blockDim.x + threadIdx.x;   // over BB*NN*DIN
  int site = i / DIN, c = i - site * DIN;
  float v;
  if (c < DD)            v = qbuf[site * DD + c];
  else if (c < 2 * DD)   v = pbuf[site * DD + (c - DD)];
  else                   v = x[i];
  out[i] = v;
}

// ---------------------------------------------------------------------------
extern "C" void kernel_launch(void* const* d_in, const int* in_sizes, int n_in,
                              void* d_out, int out_size, void* d_ws, size_t ws_size,
                              hipStream_t stream) {
  const float* x     = (const float*)d_in[0];
  const float* tfin  = (const float*)d_in[1];
  const float* W1_q  = (const float*)d_in[2];
  const float* b1_q  = (const float*)d_in[3];
  const float* W2_q  = (const float*)d_in[4];
  const float* b2_q  = (const float*)d_in[5];
  const float* W1_p  = (const float*)d_in[6];
  const float* b1_p  = (const float*)d_in[7];
  const float* W2_p  = (const float*)d_in[8];
  const float* b2_p  = (const float*)d_in[9];
  const int*   nbrs  = (const int*)d_in[10];
  float* out = (float*)d_out;

  // workspace: q (4MB) | p (4MB) | swizzled bf16 weights (~64KB)
  float* qbuf = (float*)d_ws;
  float* pbuf = qbuf + (size_t)BB * NN * DD;
  unsigned* w1q = (unsigned*)(pbuf + (size_t)BB * NN * DD);
  unsigned* w2q = w1q + 24 * 256;   // 3*8 frags * 32 lanes * 8 dwords
  unsigned* w1p = w2q + 8  * 256;   // 4*2 frags
  unsigned* w2p = w1p + 24 * 256;

  copy_in_kernel<<<(BB * NN * DD) / 256, 256, 0, stream>>>(x, qbuf, pbuf);
  swizzle_w_kernel<<<(3 * 8 * 256) / 256, 256, 0, stream>>>(W1_q, DIN,  HIDN, 8, w1q, 3 * 8 * 256);
  swizzle_w_kernel<<<(4 * 2 * 256) / 256, 256, 0, stream>>>(W2_q, HIDN, DD,   2, w2q, 4 * 2 * 256);
  swizzle_w_kernel<<<(3 * 8 * 256) / 256, 256, 0, stream>>>(W1_p, DIN,  HIDN, 8, w1p, 3 * 8 * 256);
  swizzle_w_kernel<<<(4 * 2 * 256) / 256, 256, 0, stream>>>(W2_p, HIDN, DD,   2, w2p, 4 * 2 * 256);

  const float DTSTEP = 0.25f;
  const int grid = (BB * NN) / 128;   // 256 blocks, 8 waves * 16 sites each
  float tq = 0.0f, tp = 0.0f;
  for (int k = 0; k < NSTEPS; ++k) {
    float rho = (k == 0) ? 0.5f : 1.0f;
    // q += dt_q * F(p);  cross-site coupling forces a grid-wide sync per half
    half_step_kernel<<<grid, 256, 0, stream>>>(pbuf, qbuf, x, nbrs,
                                               w1q, b1_q, w2q, b2_q, tfin, tq, rho * DTSTEP);
    tq += rho * DTSTEP;
    // p += dt_p * F(q)
    half_step_kernel<<<grid, 256, 0, stream>>>(qbuf, pbuf, x, nbrs,
                                               w1p, b1_p, w2p, b2_p, tfin, tp, DTSTEP);
    tp += DTSTEP;
  }
  finalize_kernel<<<(BB * NN * DIN) / 256, 256, 0, stream>>>(qbuf, pbuf, x, out);

  (void)in_sizes; (void)n_in; (void)out_size; (void)ws_size;
}
